// ResNet_64381559767073
// MI455X (gfx1250) — 8-bit matrix operand order repaired and checked by reading, compiled, not run on hardware
//
#include <hip/hip_runtime.h>
#include <cstdint>
#include <cstddef>


typedef __attribute__((ext_vector_type(8))) int v8i;
typedef __attribute__((ext_vector_type(4))) unsigned int u32x4;
typedef __attribute__((ext_vector_type(8))) int i32x8;
typedef __attribute__((ext_vector_type(4))) int i32x4;

#define BN_EPS 1e-5f

#ifndef __has_builtin
#define __has_builtin(x) 0
#endif
#if defined(__gfx1250__) && __has_builtin(__builtin_amdgcn_tensor_load_to_lds) && \
    __has_builtin(__builtin_amdgcn_s_wait_tensorcnt)
#define HAVE_TDM 1
#else
#define HAVE_TDM 0
#endif

#if HAVE_TDM
__device__ inline void tdm_load_b_tile(const int8_t* gaddr, uint32_t lds_off,
                                       int rows, int row_stride) {
    uint64_t ga = (uint64_t)(uintptr_t)gaddr;
    u32x4 g0;
    g0[0] = 1u;
    g0[1] = lds_off;
    g0[2] = (uint32_t)ga;
    g0[3] = (uint32_t)((ga >> 32) & 0x1FFFFFFu)
            | (2u << 30);
    i32x8 g1;
    g1[0] = 0;
    g1[1] = (int)(64u << 16);
    g1[2] = (int)((uint32_t)rows << 16);
    g1[3] = (int)(64u << 16);
    g1[4] = (int)((uint32_t)rows & 0xFFFF);
    g1[5] = row_stride;
    g1[6] = 0;
    g1[7] = 0;
    i32x4 z4 = {0, 0, 0, 0};
    i32x8 z8 = {0, 0, 0, 0, 0, 0, 0, 0};
    __builtin_amdgcn_tensor_load_to_lds(g0, g1, z4, z4, z8, 0);
}
#endif

__global__ void binarize_weights_kernel(const float* __restrict__ w,
                                        int8_t* __restrict__ out,
                                        int Cout, int Cin, int ksz) {
    int idx = blockIdx.x * blockDim.x + threadIdx.x;
    int Ktot = Cin * ksz * ksz;
    int total = Cout * Ktot;
    if (idx >= total) return;
    int cout = idx / Ktot;
    int k = idx - cout * Ktot;
    int kpos = k / Cin;
    int cin = k - kpos * Cin;
    int kh = kpos / ksz, kw = kpos - kh * ksz;
    float x = w[(((size_t)cout * Cin + cin) * ksz + kh) * ksz + kw];
    out[idx] = (x >= 0.f) ? (int8_t)1 : (int8_t)-1;
}

template <int NPW>
__global__ __launch_bounds__(128)
void binconv_wmma_kernel(const int8_t* __restrict__ act,
                         const int8_t* __restrict__ wB,
                         int Hin, int Win, int Cin, int cin_shift,
                         int Ho, int Wo, int Cout,
                         int pad, int stride, int Ktot,
                         int mode,
                         const float* __restrict__ g1, const float* __restrict__ b1,
                         const float* __restrict__ m1, const float* __restrict__ v1,
                         const float* __restrict__ g2, const float* __restrict__ b2,
                         const float* __restrict__ m2, const float* __restrict__ v2,
                         const float* __restrict__ idn,
                         int8_t* __restrict__ out_s8, float* __restrict__ out_f32) {
    __shared__ int8_t lds_a[2][32 * 64];
#if HAVE_TDM
    __shared__ int8_t lds_b[2][64 * NPW * 64];
#endif

    const int lane = threadIdx.x & 31;
    const int wave = threadIdx.x >> 5;
    const int m0 = blockIdx.x * 32;
    const int ncolbase = blockIdx.y * (64 * NPW) + wave * (16 * NPW);
    const int HW = Ho * Wo;

    const int arow = lane & 15;
    const int agrp = (lane >> 4) ? 8 : 0;
    const int bgrp = (lane >> 4) ? 16 : 0;

    const int g_row = threadIdx.x >> 2;
    const int g_seg = (threadIdx.x & 3) * 16;
    const int g_m   = m0 + g_row;
    const int g_n   = g_m / HW;
    const int g_rem = g_m - g_n * HW;
    const int g_oh  = g_rem / Wo;
    const int g_ow  = g_rem - g_oh * Wo;
    const int g_ihb = g_oh * stride - pad;
    const int g_iwb = g_ow * stride - pad;
    const int8_t* g_base = act + (size_t)g_n * Hin * Win * Cin + g_seg;

    const int8_t* b_block = wB + (size_t)(blockIdx.y * (64 * NPW)) * Ktot;
    (void)b_block;

    auto a_fetch = [&](int kc) -> int4 {
        const int kpos = kc >> cin_shift;
        const int cinb = kc & (Cin - 1);
        const int kh = kpos / 3;
        const int kw = kpos - kh * 3;
        const int ih = g_ihb + kh;
        const int iw = g_iwb + kw;
        int4 v = make_int4(0, 0, 0, 0);
        if (ih >= 0 && ih < Hin && iw >= 0 && iw < Win)
            v = *(const int4*)(g_base + ((size_t)ih * Win + iw) * Cin + cinb);
        return v;
    };

    v8i zero = {};
    v8i acc[NPW][2];
    #pragma unroll
    for (int s = 0; s < NPW; ++s) { acc[s][0] = zero; acc[s][1] = zero; }

    {
        int4 v = a_fetch(0);
        *(int4*)&lds_a[0][g_row * 64 + g_seg] = v;
#if HAVE_TDM
        if (wave == 0) {
            tdm_load_b_tile(b_block, (uint32_t)(uintptr_t)&lds_b[0][0],
                            64 * NPW, Ktot);
            __builtin_amdgcn_s_wait_tensorcnt(0);
        }
#endif
    }
    __syncthreads();

    int cur = 0;
    for (int kc = 0; kc < Ktot; kc += 64) {
        const int nxt = cur ^ 1;
        const bool more = (kc + 64) < Ktot;

        int4 aval = make_int4(0, 0, 0, 0);
        if (more) aval = a_fetch(kc + 64);
#if HAVE_TDM
        if (more && wave == 0)
            tdm_load_b_tile(b_block + kc + 64, (uint32_t)(uintptr_t)&lds_b[nxt][0],
                            64 * NPW, Ktot);
#endif

        v8i a0, a1;
        #pragma unroll
        for (int i = 0; i < 8; ++i) {
            int koff = ((i >> 2) * 32) + ((i & 3) * 4) + agrp * 2;
            a0[i] = *(const int*)&lds_a[cur][arow * 64 + koff];
            a1[i] = *(const int*)&lds_a[cur][(arow + 16) * 64 + koff];
        }
        #pragma unroll
        for (int s = 0; s < NPW; ++s) {
            v8i bf;
#if HAVE_TDM
            const int nloc = wave * (16 * NPW) + s * 16 + (lane & 15);
            const int4 q0 = *(const int4*)&lds_b[cur][nloc * 64 + bgrp];
            const int4 q1 = *(const int4*)&lds_b[cur][nloc * 64 + 32 + bgrp];
#else
            const int8_t* bt = wB + (size_t)(ncolbase + s * 16 + (lane & 15)) * Ktot + kc;
            const int4 q0 = *(const int4*)(bt + bgrp);
            const int4 q1 = *(const int4*)(bt + 32 + bgrp);
#endif
            bf[0] = q0.x; bf[1] = q0.y; bf[2] = q0.z; bf[3] = q0.w;
            bf[4] = q1.x; bf[5] = q1.y; bf[6] = q1.z; bf[7] = q1.w;
            acc[s][0] = __builtin_amdgcn_wmma_i32_16x16x64_iu8(true, a0, true, bf,
                                                               acc[s][0], false, false);
            acc[s][1] = __builtin_amdgcn_wmma_i32_16x16x64_iu8(true, a1, true, bf,
                                                               acc[s][1], false, false);
        }

        if (more) {
            *(int4*)&lds_a[nxt][g_row * 64 + g_seg] = aval;
#if HAVE_TDM
            if (wave == 0) __builtin_amdgcn_s_wait_tensorcnt(0);
#endif
        }
        __syncthreads();
        cur = nxt;
    }

    const int mbase = m0 + ((lane >> 4) ? 8 : 0);
    #pragma unroll
    for (int s = 0; s < NPW; ++s) {
        const int nc = ncolbase + s * 16 + (lane & 15);
        float a1f = g1[nc] * rsqrtf(v1[nc] + BN_EPS);
        float c1f = b1[nc] - m1[nc] * a1f;
        float a2f = 1.f, c2f = 0.f;
        if (mode == 2) { a2f = g2[nc] * rsqrtf(v2[nc] + BN_EPS); c2f = b2[nc] - m2[nc] * a2f; }
        #pragma unroll
        for (int t = 0; t < 2; ++t) {
            #pragma unroll
            for (int r = 0; r < 8; ++r) {
                const int m = mbase + t * 16 + r;
                const size_t o = (size_t)m * Cout + nc;
                float y = a1f * (float)acc[s][t][r] + c1f;
                if (mode == 0) {
                    out_s8[o] = (y >= 0.f) ? (int8_t)1 : (int8_t)-1;
                } else if (mode == 1) {
                    float z = y + idn[o];
                    z = fminf(fmaxf(z, -1.f), 1.f);
                    out_f32[o] = z;
                    out_s8[o] = (z >= 0.f) ? (int8_t)1 : (int8_t)-1;
                } else {
                    out_f32[o] = a2f * y + c2f;
                }
            }
        }
    }
}

__global__ void stem_conv_kernel(const float* __restrict__ x,
                                 const float* __restrict__ w,
                                 float* __restrict__ y) {
    int idx = blockIdx.x * blockDim.x + threadIdx.x;
    const int total = 32 * 112 * 112 * 64;
    if (idx >= total) return;
    int co = idx & 63;
    int t = idx >> 6;
    int ow = t % 112; t /= 112;
    int oh = t % 112; int n = t / 112;
    float s = 0.f;
    for (int ci = 0; ci < 3; ++ci)
        for (int kh = 0; kh < 7; ++kh) {
            int ih = oh * 2 - 3 + kh;
            if (ih < 0 || ih >= 224) continue;
            for (int kw = 0; kw < 7; ++kw) {
                int iw = ow * 2 - 3 + kw;
                if (iw < 0 || iw >= 224) continue;
                s += x[((size_t)(n * 3 + ci) * 224 + ih) * 224 + iw] *
                     w[((size_t)(co * 3 + ci) * 7 + kh) * 7 + kw];
            }
        }
    y[idx] = s;
}

__global__ void pool_bn_kernel(const float* __restrict__ y,
                               const float* __restrict__ g, const float* __restrict__ b,
                               const float* __restrict__ mm, const float* __restrict__ vv,
                               float* __restrict__ xf, int8_t* __restrict__ xs) {
    int idx = blockIdx.x * blockDim.x + threadIdx.x;
    const int total = 32 * 56 * 56 * 64;
    if (idx >= total) return;
    int c = idx & 63;
    int t = idx >> 6;
    int ow = t % 56; t /= 56;
    int oh = t % 56; int n = t / 56;
    float mx = -3.402823e38f;
    for (int kh = 0; kh < 3; ++kh) {
        int ih = oh * 2 - 1 + kh;
        if (ih < 0 || ih >= 112) continue;
        for (int kw = 0; kw < 3; ++kw) {
            int iw = ow * 2 - 1 + kw;
            if (iw < 0 || iw >= 112) continue;
            mx = fmaxf(mx, y[(((size_t)n * 112 + ih) * 112 + iw) * 64 + c]);
        }
    }
    float a = g[c] * rsqrtf(vv[c] + BN_EPS);
    float z = (mx - mm[c]) * a + b[c];
    z = fminf(fmaxf(z, -1.f), 1.f);
    xf[idx] = z;
    xs[idx] = (z >= 0.f) ? (int8_t)1 : (int8_t)-1;
}

__global__ void head_pool_kernel(const float* __restrict__ x,
                                 const float* __restrict__ g, const float* __restrict__ b,
                                 const float* __restrict__ mm, const float* __restrict__ vv,
                                 float* __restrict__ feat) {
    int idx = blockIdx.x * blockDim.x + threadIdx.x;
    if (idx >= 32 * 512) return;
    int c = idx & 511;
    int n = idx >> 9;
    float s = 0.f;
    for (int h = 0; h < 7; ++h)
        for (int w = 0; w < 7; ++w)
            s += x[(((size_t)n * 7 + h) * 7 + w) * 512 + c];
    s *= (1.f / 49.f);
    float a = g[c] * rsqrtf(vv[c] + BN_EPS);
    float z = (s - mm[c]) * a + b[c];
    feat[idx] = fminf(fmaxf(z, -1.f), 1.f);
}

__global__ void fc_kernel(const float* __restrict__ feat,
                          const float* __restrict__ W,
                          const float* __restrict__ bias,
                          float* __restrict__ out) {
    int idx = blockIdx.x * blockDim.x + threadIdx.x;
    if (idx >= 32 * 128) return;
    int o = idx & 127;
    int n = idx >> 7;
    float s = bias[o];
    for (int k = 0; k < 512; ++k)
        s += feat[n * 512 + k] * W[o * 512 + k];
    out[idx] = s;
}

static void run_conv(hipStream_t stream, const int8_t* act, const int8_t* wB,
                     int Hin, int Cin, int Ho, int Cout, int pad, int stride,
                     int Ktot, int mode,
                     const float* g1, const float* b1, const float* m1, const float* v1,
                     const float* g2, const float* b2, const float* m2, const float* v2,
                     const float* idn, int8_t* out_s8, float* out_f32) {
    const int M = 32 * Ho * Ho;
    int cs = 0; while (!((Cin >> cs) & 1)) ++cs;
    if (Cout % 128 == 0) {
        dim3 grid(M / 32, Cout / 128);
        binconv_wmma_kernel<2><<<grid, 128, 0, stream>>>(
            act, wB, Hin, Hin, Cin, cs, Ho, Ho, Cout, pad, stride, Ktot, mode,
            g1, b1, m1, v1, g2, b2, m2, v2, idn, out_s8, out_f32);
    } else {
        dim3 grid(M / 32, Cout / 64);
        binconv_wmma_kernel<1><<<grid, 128, 0, stream>>>(
            act, wB, Hin, Hin, Cin, cs, Ho, Ho, Cout, pad, stride, Ktot, mode,
            g1, b1, m1, v1, g2, b2, m2, v2, idn, out_s8, out_f32);
    }
}

extern "C" void kernel_launch(void* const* d_in, const int* in_sizes, int n_in,
                              void* d_out, int out_size, void* d_ws, size_t ws_size,
                              hipStream_t stream) {
    (void)in_sizes; (void)n_in; (void)out_size; (void)ws_size;

    struct Bn { int b, g, m, v; };
    const bool hasds[8] = {false,false,true,false,true,false,true,false};
    Bn blk_bn1[8], blk_bn2[8], blk_bn3[8], blk_dsbn[8];
    int blk_c1[8], blk_c2[8], blk_ds[8];
    int p = 0;
    auto take_bn = [&p]() { Bn r{p, p + 1, p + 2, p + 3}; p += 4; return r; };
    for (int i = 0; i < 8; ++i) {
        blk_bn1[i] = take_bn();
        blk_bn2[i] = take_bn();
        if (hasds[i]) {
            blk_bn3[i] = take_bn();
            blk_c1[i] = p++; blk_c2[i] = p++;
            blk_dsbn[i] = take_bn();
            blk_ds[i] = p++;
        } else {
            blk_c1[i] = p++; blk_c2[i] = p++;
        }
    }
    Bn stem_bn = take_bn();
    Bn head_bn = take_bn();
    int fc_b_i = p++, fc_w_i = p++, stem_w_i = p++, x_i = p++;
    auto F = [&](int i) { return (const float*)d_in[i]; };

    char* ws = (char*)d_ws;
    size_t off = 0;
    auto alloc = [&](size_t bytes) -> char* {
        char* ptr = ws + off;
        off = (off + bytes + 255) & ~(size_t)255;
        return ptr;
    };
    const int NB = 32;
    float*  stem_y = (float*) alloc((size_t)NB * 112 * 112 * 64 * 4);
    float*  xf[2]  = {(float*)alloc((size_t)NB*56*56*64*4), (float*)alloc((size_t)NB*56*56*64*4)};
    int8_t* xs[2]  = {(int8_t*)alloc((size_t)NB*56*56*64),  (int8_t*)alloc((size_t)NB*56*56*64)};
    int8_t* t_s8   = (int8_t*)alloc((size_t)NB * 56 * 56 * 64);
    float*  idn_f  = (float*) alloc((size_t)NB * 28 * 28 * 128 * 4);
    float*  feat   = (float*) alloc((size_t)NB * 512 * 4);

    const int cfg_cin[8]    = {64, 64, 64, 128, 128, 256, 256, 512};
    const int cfg_cout[8]   = {64, 64, 128, 128, 256, 256, 512, 512};
    const int cfg_stride[8] = {1, 1, 2, 1, 2, 1, 2, 1};
    int8_t *w1b[8], *w2b[8], *wdb[8];
    for (int i = 0; i < 8; ++i) {
        w1b[i] = (int8_t*)alloc((size_t)cfg_cout[i] * cfg_cin[i] * 9);
        w2b[i] = (int8_t*)alloc((size_t)cfg_cout[i] * cfg_cout[i] * 9);
        wdb[i] = hasds[i] ? (int8_t*)alloc((size_t)cfg_cout[i] * cfg_cin[i]) : nullptr;
    }

    auto binw = [&](const float* w, int8_t* out, int Cout, int Cin, int ksz) {
        int total = Cout * Cin * ksz * ksz;
        binarize_weights_kernel<<<(total + 255) / 256, 256, 0, stream>>>(w, out, Cout, Cin, ksz);
    };
    for (int i = 0; i < 8; ++i) {
        binw(F(blk_c1[i]), w1b[i], cfg_cout[i], cfg_cin[i], 3);
        binw(F(blk_c2[i]), w2b[i], cfg_cout[i], cfg_cout[i], 3);
        if (hasds[i]) binw(F(blk_ds[i]), wdb[i], cfg_cout[i], cfg_cin[i], 1);
    }

    {
        int total = NB * 112 * 112 * 64;
        stem_conv_kernel<<<(total + 255) / 256, 256, 0, stream>>>(F(x_i), F(stem_w_i), stem_y);
        int total2 = NB * 56 * 56 * 64;
        pool_bn_kernel<<<(total2 + 255) / 256, 256, 0, stream>>>(
            stem_y, F(stem_bn.g), F(stem_bn.b), F(stem_bn.m), F(stem_bn.v), xf[0], xs[0]);
    }

    int cur = 0, Hin = 56;
    for (int i = 0; i < 8; ++i) {
        const int Cin = cfg_cin[i], Cout = cfg_cout[i], s = cfg_stride[i];
        const int Ho = Hin / s;

        const float* idn;
        if (hasds[i]) {
            run_conv(stream, xs[cur], wdb[i], Hin, Cin, Ho, Cout, 0, s, Cin, 2,
                     F(blk_dsbn[i].g), F(blk_dsbn[i].b), F(blk_dsbn[i].m), F(blk_dsbn[i].v),
                     F(blk_bn3[i].g),  F(blk_bn3[i].b),  F(blk_bn3[i].m),  F(blk_bn3[i].v),
                     nullptr, nullptr, idn_f);
            idn = idn_f;
        } else {
            idn = xf[cur];
        }

        run_conv(stream, xs[cur], w1b[i], Hin, Cin, Ho, Cout, 1, s, Cin * 9, 0,
                 F(blk_bn1[i].g), F(blk_bn1[i].b), F(blk_bn1[i].m), F(blk_bn1[i].v),
                 nullptr, nullptr, nullptr, nullptr, nullptr, t_s8, nullptr);

        run_conv(stream, t_s8, w2b[i], Ho, Cout, Ho, Cout, 1, 1, Cout * 9, 1,
                 F(blk_bn2[i].g), F(blk_bn2[i].b), F(blk_bn2[i].m), F(blk_bn2[i].v),
                 nullptr, nullptr, nullptr, nullptr, idn, xs[1 - cur], xf[1 - cur]);

        cur = 1 - cur;
        Hin = Ho;
    }

    head_pool_kernel<<<(NB * 512 + 255) / 256, 256, 0, stream>>>(
        xf[cur], F(head_bn.g), F(head_bn.b), F(head_bn.m), F(head_bn.v), feat);
    fc_kernel<<<(NB * 128 + 255) / 256, 256, 0, stream>>>(
        feat, F(fc_w_i), F(fc_b_i), (float*)d_out);
}
